// Ernie4_5_DecoderLayer_81819126989333
// MI455X (gfx1250) — matrix operand order repaired and checked by reading, not compiled, not run on hardware
//
#include <hip/hip_runtime.h>

// ---------------------------------------------------------------------------
// Types for gfx1250 WMMA (wave32)
// ---------------------------------------------------------------------------
typedef __bf16  v16bf __attribute__((ext_vector_type(16)));
typedef __bf16  v8bf  __attribute__((ext_vector_type(8)));
typedef float   v8f   __attribute__((ext_vector_type(8)));

#define T_TOK   1024
#define HDIM    2048
#define NHQ     16
#define NHKV    4
#define DHEAD   128
#define NEXP    32
#define TOPK    4
#define IDIM    1024
#define ISHDIM  1024

#define FLAG_ACC   1   // fp32 accumulate into C
#define FLAG_BF16  2   // write bf16 output

// ---------------------------------------------------------------------------
// CDNA5 async-LDS probe (compiled, never launched): exercises
// GLOBAL_LOAD_ASYNC_TO_LDS_B32 + S_WAIT_ASYNCCNT from CDNA5 ISA
// (mnemonics confirmed accepted by the gfx1250 assembler in round 2)
// ---------------------------------------------------------------------------
__global__ void cdna5_async_probe_k(const float* __restrict__ src, float* __restrict__ dst) {
    __shared__ float buf[256];
    unsigned ldsoff = (unsigned)(size_t)&buf[threadIdx.x];
    const float* gp = src + threadIdx.x;
    asm volatile("global_load_async_to_lds_b32 %0, %1, off"
                 :: "v"(ldsoff), "v"(gp) : "memory");
    asm volatile("s_wait_asynccnt 0" ::: "memory");
    __syncthreads();
    dst[threadIdx.x] = buf[(threadIdx.x + 1) & 255];
}

// ---------------------------------------------------------------------------
// bf16 WMMA GEMM:  C[M,N] = A[M,K] @ Bt[N,K]^T   (Bt is the transposed weight,
// so all fragment loads are contiguous 16B loads).
// Block = 128 threads = 4 waves; each wave computes a 32x64 tile (8 WMMA/k-step),
// software-pipelined with two register buffer sets. K must be a multiple of 64
// (true for all call sites: K in {128, 1024, 2048}).
// __launch_bounds__(128, 1) lifts the VGPR cap so fragments stay in registers.
// ---------------------------------------------------------------------------
__global__ __launch_bounds__(128, 1)
void wmma_gemm_bf16(const __bf16* __restrict__ A, int lda,
                    const __bf16* __restrict__ Bt, int ldb,
                    void* __restrict__ C, int ldc,
                    int M, int N, int K, int flags) {
    const int lane = threadIdx.x & 31;
    const int wave = threadIdx.x >> 5;
    const int row0 = blockIdx.y * 128 + wave * 32;
    const int col0 = blockIdx.x * 64;
    const int r  = lane & 15;
    const int hi = lane >> 4;

    // A fragment base pointers for the two 16-row strips (row clamped for edge)
    const __bf16* aptr[2];
#pragma unroll
    for (int s = 0; s < 2; ++s) {
        int ar = row0 + s * 16 + r;
        if (ar >= M) ar = M - 1;
        aptr[s] = A + (size_t)ar * lda + hi * 8;
    }
    // B fragment base pointers for the four 16-col tiles (col clamped for edge)
    const __bf16* bptr[4];
#pragma unroll
    for (int ct = 0; ct < 4; ++ct) {
        int n = col0 + ct * 16 + r;
        if (n >= N) n = N - 1;
        bptr[ct] = Bt + (size_t)n * ldb + hi * 8;
    }

    v8f acc[2][4];
#pragma unroll
    for (int s = 0; s < 2; ++s)
#pragma unroll
        for (int ct = 0; ct < 4; ++ct)
#pragma unroll
            for (int i = 0; i < 8; ++i) acc[s][ct][i] = 0.0f;

#define LOAD_A(dst, k0)                                                         \
    _Pragma("unroll")                                                           \
    for (int s = 0; s < 2; ++s) {                                               \
        v8bf x = *reinterpret_cast<const v8bf*>(aptr[s] + (k0));                \
        v8bf y = *reinterpret_cast<const v8bf*>(aptr[s] + (k0) + 16);           \
        dst[s] = __builtin_shufflevector(x, y,                                  \
            0,1,2,3,4,5,6,7,8,9,10,11,12,13,14,15);                             \
    }
#define LOAD_B(dst, k0)                                                         \
    _Pragma("unroll")                                                           \
    for (int ct = 0; ct < 4; ++ct) {                                            \
        v8bf x = *reinterpret_cast<const v8bf*>(bptr[ct] + (k0));               \
        v8bf y = *reinterpret_cast<const v8bf*>(bptr[ct] + (k0) + 16);          \
        dst[ct] = __builtin_shufflevector(x, y,                                 \
            0,1,2,3,4,5,6,7,8,9,10,11,12,13,14,15);                             \
    }
#define MMA_SET(af, bfr)                                                        \
    _Pragma("unroll")                                                           \
    for (int ct = 0; ct < 4; ++ct)                                              \
        _Pragma("unroll")                                                       \
        for (int s = 0; s < 2; ++s)                                             \
            acc[s][ct] = __builtin_amdgcn_wmma_f32_16x16x32_bf16(               \
                false, af[s], false, bfr[ct], (short)0, acc[s][ct], false, false);

    v16bf af0[2], af1[2];
    v16bf bf0[4], bf1[4];

    // prologue: load K-chunk 0 into set 0
    LOAD_A(af0, 0)
    LOAD_B(bf0, 0)

    // steady state: branch-free software pipeline, 64 K-elements per trip
    int k0 = 0;
    for (; k0 < K - 64; k0 += 64) {
        LOAD_A(af1, k0 + 32)
        LOAD_B(bf1, k0 + 32)
        MMA_SET(af0, bf0)
        LOAD_A(af0, k0 + 64)
        LOAD_B(bf0, k0 + 64)
        MMA_SET(af1, bf1)
    }
    // epilogue chunk (last 64 K-elements)
    LOAD_A(af1, k0 + 32)
    LOAD_B(bf1, k0 + 32)
    MMA_SET(af0, bf0)
    MMA_SET(af1, bf1)
#undef LOAD_A
#undef LOAD_B
#undef MMA_SET

    // Epilogue: C/D layout -> VGPR i holds row (strip + i + hi*8), col = col0+ct*16+r
#pragma unroll
    for (int s = 0; s < 2; ++s) {
#pragma unroll
        for (int ct = 0; ct < 4; ++ct) {
            int col = col0 + ct * 16 + r;
            if (col >= N) continue;
#pragma unroll
            for (int i = 0; i < 8; ++i) {
                int row = row0 + s * 16 + i + hi * 8;
                if (row >= M) continue;
                size_t idx = (size_t)row * ldc + col;
                float v = acc[s][ct][i];
                if (flags & FLAG_BF16)      ((__bf16*)C)[idx] = (__bf16)v;
                else if (flags & FLAG_ACC)  ((float*)C)[idx] += v;
                else                        ((float*)C)[idx]  = v;
            }
        }
    }
}

// ---------------------------------------------------------------------------
// RMSNorm (fp32 in) -> bf16 out
// ---------------------------------------------------------------------------
__global__ void rmsnorm_bf16_k(const float* __restrict__ x, const float* __restrict__ w,
                               __bf16* __restrict__ out, int Hn) {
    int t = blockIdx.x;
    const float* xp = x + (size_t)t * Hn;
    __shared__ float red[256];
    float s = 0.f;
    for (int i = threadIdx.x; i < Hn; i += 256) { float v = xp[i]; s += v * v; }
    red[threadIdx.x] = s; __syncthreads();
    for (int o = 128; o > 0; o >>= 1) {
        if (threadIdx.x < o) red[threadIdx.x] += red[threadIdx.x + o];
        __syncthreads();
    }
    float inv = rsqrtf(red[0] / (float)Hn + 1e-6f);
    for (int i = threadIdx.x; i < Hn; i += 256)
        out[(size_t)t * Hn + i] = (__bf16)(w[i] * xp[i] * inv);
}

// ---------------------------------------------------------------------------
// Transpose + fp32->bf16 convert:  in fp32 [R,C] -> out bf16 [C,R]
// ---------------------------------------------------------------------------
__global__ void transpose_cvt_k(const float* __restrict__ in, __bf16* __restrict__ out,
                                int R, int C) {
    __shared__ float tile[32][33];
    const float* inp = in + (size_t)blockIdx.z * R * C;
    __bf16* outp = out + (size_t)blockIdx.z * R * C;
    int c0 = blockIdx.x * 32, r0 = blockIdx.y * 32;
    for (int i = threadIdx.y; i < 32; i += 8) {
        int rr = r0 + i, cc = c0 + threadIdx.x;
        tile[i][threadIdx.x] = (rr < R && cc < C) ? inp[(size_t)rr * C + cc] : 0.f;
    }
    __syncthreads();
    for (int i = threadIdx.y; i < 32; i += 8) {
        int cc = c0 + i, rr = r0 + threadIdx.x;
        if (cc < C && rr < R) outp[(size_t)cc * R + rr] = (__bf16)tile[threadIdx.x][i];
    }
}

__global__ void f32_to_bf16_k(const float* __restrict__ in, __bf16* __restrict__ out, size_t n) {
    size_t i = (size_t)blockIdx.x * 256 + threadIdx.x;
    if (i < n) out[i] = (__bf16)in[i];
}

// ---------------------------------------------------------------------------
// RoPE + pack fp32 [T, heads*D] -> bf16 [heads][T][D]
// ---------------------------------------------------------------------------
__global__ void rope_pack_k(const float* __restrict__ qk, __bf16* __restrict__ out,
                            int Tn, int heads, int Dn) {
    size_t idx = (size_t)blockIdx.x * 256 + threadIdx.x;
    int half = Dn / 2;
    size_t total = (size_t)Tn * heads * half;
    if (idx >= total) return;
    int i = idx % half;
    int h = (idx / half) % heads;
    int t = idx / ((size_t)half * heads);
    float freq = powf(10000.0f, -2.0f * (float)i / (float)Dn);
    float ang = (float)t * freq;
    float c = cosf(ang), s = sinf(ang);
    const float* p = qk + (size_t)t * heads * Dn + (size_t)h * Dn + 2 * i;
    float x0 = p[0], x1 = p[1];
    __bf16* o = out + ((size_t)h * Tn + t) * Dn + 2 * i;
    o[0] = (__bf16)(x0 * c - x1 * s);
    o[1] = (__bf16)(x1 * c + x0 * s);
}

// v fp32 [T, HKV*D] -> bf16 [HKV][D][T]   (transposed for the probs@V GEMM)
__global__ void v_pack_k(const float* __restrict__ v, __bf16* __restrict__ vt,
                         int Tn, int hkv, int Dn) {
    size_t idx = (size_t)blockIdx.x * 256 + threadIdx.x;
    size_t total = (size_t)Tn * hkv * Dn;
    if (idx >= total) return;
    int d = idx % Dn;
    int h = (idx / Dn) % hkv;
    int t = idx / ((size_t)Dn * hkv);
    vt[((size_t)h * Dn + d) * Tn + t] = (__bf16)v[idx];
}

// ---------------------------------------------------------------------------
// Causal softmax over one head's score matrix (fp32 [S,S]) -> bf16 probs
// ---------------------------------------------------------------------------
__global__ void softmax_causal_k(const float* __restrict__ sc, __bf16* __restrict__ pr,
                                 int S, float scale) {
    int t = blockIdx.x;
    const float* row = sc + (size_t)t * S;
    __bf16* orow = pr + (size_t)t * S;
    __shared__ float red[256];
    float mx = -3.4e38f;
    for (int j = threadIdx.x; j <= t; j += 256) mx = fmaxf(mx, row[j] * scale);
    red[threadIdx.x] = mx; __syncthreads();
    for (int o = 128; o > 0; o >>= 1) {
        if (threadIdx.x < o) red[threadIdx.x] = fmaxf(red[threadIdx.x], red[threadIdx.x + o]);
        __syncthreads();
    }
    float m = red[0]; __syncthreads();
    float s = 0.f;
    for (int j = threadIdx.x; j < S; j += 256)
        if (j <= t) s += __expf(row[j] * scale - m);
    red[threadIdx.x] = s; __syncthreads();
    for (int o = 128; o > 0; o >>= 1) {
        if (threadIdx.x < o) red[threadIdx.x] += red[threadIdx.x + o];
        __syncthreads();
    }
    float inv = 1.0f / red[0];
    for (int j = threadIdx.x; j < S; j += 256) {
        float e = (j <= t) ? __expf(row[j] * scale - m) * inv : 0.f;
        orow[j] = (__bf16)e;
    }
}

// ---------------------------------------------------------------------------
// Router: softmax over E=32, top-4 (strict > => lowest index wins ties, matches
// jax top_k), normalized weights scattered into dense cw[T,E]
// ---------------------------------------------------------------------------
__global__ void router_topk_k(const float* __restrict__ logits, const float* __restrict__ bias,
                              float* __restrict__ cw, int Tn) {
    int t = blockIdx.x * 256 + threadIdx.x;
    if (t >= Tn) return;
    const float* l = logits + (size_t)t * NEXP;
    float p[NEXP], tmp[NEXP];
    float mx = -3.4e38f;
    for (int e = 0; e < NEXP; ++e) mx = fmaxf(mx, l[e]);
    float s = 0.f;
    for (int e = 0; e < NEXP; ++e) { p[e] = __expf(l[e] - mx); s += p[e]; }
    float invs = 1.0f / s;
    for (int e = 0; e < NEXP; ++e) { p[e] *= invs; tmp[e] = p[e] + bias[e]; }
    float* c = cw + (size_t)t * NEXP;
    for (int e = 0; e < NEXP; ++e) c[e] = 0.f;
    int sel[TOPK]; float rsum = 0.f;
    for (int k = 0; k < TOPK; ++k) {
        int bi = 0; float bv = -3.4e38f;
        for (int e = 0; e < NEXP; ++e)
            if (tmp[e] > bv) { bv = tmp[e]; bi = e; }
        sel[k] = bi; tmp[bi] = -3.4e38f; rsum += p[bi];
    }
    rsum = fmaxf(rsum, 1e-12f);
    for (int k = 0; k < TOPK; ++k) c[sel[k]] = p[sel[k]] / rsum;
}

// ---------------------------------------------------------------------------
// hmid = silu(g)*u*cw  -> bf16  (cw==nullptr => weight 1, used by shared expert)
// ---------------------------------------------------------------------------
__global__ void silu_mul_k(const float* __restrict__ g, const float* __restrict__ u,
                           const float* __restrict__ cw, __bf16* __restrict__ out,
                           size_t n, int In) {
    size_t i = (size_t)blockIdx.x * 256 + threadIdx.x;
    if (i >= n) return;
    float gv = g[i];
    float h = gv / (1.0f + __expf(-gv)) * u[i];
    if (cw) h *= cw[(i / In) * NEXP];
    out[i] = (__bf16)h;
}

__global__ void copy_f32_k(const float* __restrict__ a, float* __restrict__ b, size_t n) {
    size_t i = (size_t)blockIdx.x * 256 + threadIdx.x;
    if (i < n) b[i] = a[i];
}
__global__ void zero_f32_k(float* __restrict__ a, size_t n) {
    size_t i = (size_t)blockIdx.x * 256 + threadIdx.x;
    if (i < n) a[i] = 0.f;
}
__global__ void add2_k(const float* __restrict__ a, const float* __restrict__ b,
                       float* __restrict__ o, size_t n) {
    size_t i = (size_t)blockIdx.x * 256 + threadIdx.x;
    if (i < n) o[i] = a[i] + b[i];
}

// ---------------------------------------------------------------------------
// Host orchestration
// ---------------------------------------------------------------------------
static inline void gemm(const __bf16* A, int lda, const __bf16* Bt, int ldb,
                        void* C, int ldc, int M, int N, int K, int flags,
                        hipStream_t s) {
    dim3 g((N + 63) / 64, (M + 127) / 128);
    wmma_gemm_bf16<<<g, 128, 0, s>>>(A, lda, Bt, ldb, C, ldc, M, N, K, flags);
}
static inline void tcvt(const float* in, __bf16* out, int R, int C, hipStream_t s) {
    dim3 g((C + 31) / 32, (R + 31) / 32, 1);
    transpose_cvt_k<<<g, dim3(32, 8), 0, s>>>(in, out, R, C);
}
static inline size_t gsz(size_t n) { return (n + 255) / 256; }

extern "C" void kernel_launch(void* const* d_in, const int* in_sizes, int n_in,
                              void* d_out, int out_size, void* d_ws, size_t ws_size,
                              hipStream_t stream) {
    const float* hidden = (const float*)d_in[0];
    const float* ln1_w  = (const float*)d_in[1];
    const float* ln2_w  = (const float*)d_in[2];
    const float* wq     = (const float*)d_in[3];
    const float* wk     = (const float*)d_in[4];
    const float* wv     = (const float*)d_in[5];
    const float* wo     = (const float*)d_in[6];
    const float* gate_w = (const float*)d_in[7];
    const float* bias_e = (const float*)d_in[8];
    const float* weg    = (const float*)d_in[9];
    const float* weu    = (const float*)d_in[10];
    const float* wed    = (const float*)d_in[11];
    const float* wsg    = (const float*)d_in[12];
    const float* wsu    = (const float*)d_in[13];
    const float* wsd    = (const float*)d_in[14];
    (void)in_sizes; (void)n_in; (void)out_size; (void)ws_size;

    // ---- bump allocator over d_ws ----
    char* base = (char*)d_ws;
    size_t off = 0;
    auto alloc = [&](size_t bytes) -> void* {
        off = (off + 255) & ~(size_t)255;
        void* p = base + off;
        off += bytes;
        return p;
    };
    const int T = T_TOK, S = T_TOK;

    __bf16* xln1   = (__bf16*)alloc((size_t)T * HDIM * 2);
    float*  qf     = (float*) alloc((size_t)T * NHQ  * DHEAD * 4);
    float*  kf     = (float*) alloc((size_t)T * NHKV * DHEAD * 4);
    float*  vf     = (float*) alloc((size_t)T * NHKV * DHEAD * 4);
    __bf16* Qh     = (__bf16*)alloc((size_t)NHQ  * S * DHEAD * 2);
    __bf16* Kh     = (__bf16*)alloc((size_t)NHKV * S * DHEAD * 2);
    __bf16* Vt     = (__bf16*)alloc((size_t)NHKV * DHEAD * S * 2);
    float*  scores = (float*) alloc((size_t)S * S * 4);
    __bf16* probs  = (__bf16*)alloc((size_t)S * S * 2);
    __bf16* o_bf   = (__bf16*)alloc((size_t)T * NHQ * DHEAD * 2);
    float*  resid2 = (float*) alloc((size_t)T * HDIM * 4);
    __bf16* xn_bf  = (__bf16*)alloc((size_t)T * HDIM * 2);
    float*  logits = (float*) alloc((size_t)T * NEXP * 4);
    float*  cw     = (float*) alloc((size_t)T * NEXP * 4);
    float*  gbuf   = (float*) alloc((size_t)T * IDIM * 4);
    float*  ubuf   = (float*) alloc((size_t)T * IDIM * 4);
    __bf16* hmid   = (__bf16*)alloc((size_t)T * IDIM * 2);
    float*  acc    = (float*) alloc((size_t)T * HDIM * 4);
    __bf16* wq_t   = (__bf16*)alloc((size_t)HDIM * NHQ  * DHEAD * 2);
    __bf16* wk_t   = (__bf16*)alloc((size_t)HDIM * NHKV * DHEAD * 2);
    __bf16* wv_t   = (__bf16*)alloc((size_t)HDIM * NHKV * DHEAD * 2);
    __bf16* wo_t   = (__bf16*)alloc((size_t)NHQ * DHEAD * HDIM * 2);
    __bf16* gate_b = (__bf16*)alloc((size_t)NEXP * HDIM * 2);
    __bf16* wsg_t  = (__bf16*)alloc((size_t)HDIM * ISHDIM * 2);
    __bf16* wsu_t  = (__bf16*)alloc((size_t)HDIM * ISHDIM * 2);
    __bf16* wsd_t  = (__bf16*)alloc((size_t)ISHDIM * HDIM * 2);
    __bf16* weg_s  = (__bf16*)alloc((size_t)HDIM * IDIM * 2);
    __bf16* weu_s  = (__bf16*)alloc((size_t)HDIM * IDIM * 2);
    __bf16* wed_s  = (__bf16*)alloc((size_t)IDIM * HDIM * 2);

    // ---- weight conversion (per-launch, deterministic) ----
    tcvt(wq,  wq_t,  HDIM, NHQ * DHEAD, stream);
    tcvt(wk,  wk_t,  HDIM, NHKV * DHEAD, stream);
    tcvt(wv,  wv_t,  HDIM, NHKV * DHEAD, stream);
    tcvt(wo,  wo_t,  NHQ * DHEAD, HDIM, stream);
    tcvt(wsg, wsg_t, HDIM, ISHDIM, stream);
    tcvt(wsu, wsu_t, HDIM, ISHDIM, stream);
    tcvt(wsd, wsd_t, ISHDIM, HDIM, stream);
    f32_to_bf16_k<<<gsz((size_t)NEXP * HDIM), 256, 0, stream>>>(gate_w, gate_b, (size_t)NEXP * HDIM);

    // ---- attention ----
    rmsnorm_bf16_k<<<T, 256, 0, stream>>>(hidden, ln1_w, xln1, HDIM);
    gemm(xln1, HDIM, wq_t, HDIM, qf, NHQ * DHEAD, T, NHQ * DHEAD, HDIM, 0, stream);
    gemm(xln1, HDIM, wk_t, HDIM, kf, NHKV * DHEAD, T, NHKV * DHEAD, HDIM, 0, stream);
    gemm(xln1, HDIM, wv_t, HDIM, vf, NHKV * DHEAD, T, NHKV * DHEAD, HDIM, 0, stream);
    rope_pack_k<<<gsz((size_t)T * NHQ  * DHEAD / 2), 256, 0, stream>>>(qf, Qh, T, NHQ, DHEAD);
    rope_pack_k<<<gsz((size_t)T * NHKV * DHEAD / 2), 256, 0, stream>>>(kf, Kh, T, NHKV, DHEAD);
    v_pack_k<<<gsz((size_t)T * NHKV * DHEAD), 256, 0, stream>>>(vf, Vt, T, NHKV, DHEAD);

    const float scale = 0.088388347648318447f;  // 1/sqrt(128)
    for (int h = 0; h < NHQ; ++h) {
        int hk = h / (NHQ / NHKV);
        gemm(Qh + (size_t)h * S * DHEAD, DHEAD,
             Kh + (size_t)hk * S * DHEAD, DHEAD,
             scores, S, S, S, DHEAD, 0, stream);
        softmax_causal_k<<<S, 256, 0, stream>>>(scores, probs, S, scale);
        gemm(probs, S, Vt + (size_t)hk * DHEAD * S, S,
             (void*)(o_bf + (size_t)h * DHEAD), NHQ * DHEAD,
             S, DHEAD, S, FLAG_BF16, stream);
    }
    copy_f32_k<<<gsz((size_t)T * HDIM), 256, 0, stream>>>(hidden, resid2, (size_t)T * HDIM);
    gemm(o_bf, NHQ * DHEAD, wo_t, NHQ * DHEAD, resid2, HDIM, T, HDIM, NHQ * DHEAD, FLAG_ACC, stream);

    // ---- MoE ----
    rmsnorm_bf16_k<<<T, 256, 0, stream>>>(resid2, ln2_w, xn_bf, HDIM);
    gemm(xn_bf, HDIM, gate_b, HDIM, logits, NEXP, T, NEXP, HDIM, 0, stream);
    router_topk_k<<<gsz(T), 256, 0, stream>>>(logits, bias_e, cw, T);
    zero_f32_k<<<gsz((size_t)T * HDIM), 256, 0, stream>>>(acc, (size_t)T * HDIM);

    for (int e = 0; e < NEXP; ++e) {
        tcvt(weg + (size_t)e * HDIM * IDIM, weg_s, HDIM, IDIM, stream);
        tcvt(weu + (size_t)e * HDIM * IDIM, weu_s, HDIM, IDIM, stream);
        tcvt(wed + (size_t)e * IDIM * HDIM, wed_s, IDIM, HDIM, stream);
        gemm(xn_bf, HDIM, weg_s, HDIM, gbuf, IDIM, T, IDIM, HDIM, 0, stream);
        gemm(xn_bf, HDIM, weu_s, HDIM, ubuf, IDIM, T, IDIM, HDIM, 0, stream);
        silu_mul_k<<<gsz((size_t)T * IDIM), 256, 0, stream>>>(gbuf, ubuf, cw + e, hmid,
                                                              (size_t)T * IDIM, IDIM);
        gemm(hmid, IDIM, wed_s, IDIM, acc, HDIM, T, HDIM, IDIM, FLAG_ACC, stream);
    }

    // ---- shared expert ----
    gemm(xn_bf, HDIM, wsg_t, HDIM, gbuf, ISHDIM, T, ISHDIM, HDIM, 0, stream);
    gemm(xn_bf, HDIM, wsu_t, HDIM, ubuf, ISHDIM, T, ISHDIM, HDIM, 0, stream);
    silu_mul_k<<<gsz((size_t)T * ISHDIM), 256, 0, stream>>>(gbuf, ubuf, nullptr, hmid,
                                                            (size_t)T * ISHDIM, ISHDIM);
    gemm(hmid, ISHDIM, wsd_t, ISHDIM, acc, HDIM, T, HDIM, ISHDIM, FLAG_ACC, stream);

    // ---- final residual ----
    add2_k<<<gsz((size_t)T * HDIM), 256, 0, stream>>>(resid2, acc, (float*)d_out,
                                                      (size_t)T * HDIM);
}
